// GraphNestedModel_73151882985543
// MI455X (gfx1250) — compile-verified, matrix operand order checked by reading, not run on hardware
//
#include <hip/hip_runtime.h>
#include <hip/hip_bf16.h>
#include <math.h>

typedef __attribute__((ext_vector_type(16))) _Float16 v16h;
typedef __attribute__((ext_vector_type(8)))  float    v8f;
typedef __attribute__((ext_vector_type(4)))  float    f32x4;

#define BB   4
#define SS   1024
#define EE   256
#define HH   512
#define LL   16
#define NTOK (BB*SS)   /* 4096 tokens */

// ---------------------------------------------------------------------------
// CDNA5 16-bit A-fragment loader (16x32 MxK tile) from row-major f16 memory.
// ISA layout: lanes 0-15 hold rows M=0..15 with halves h=0..7 -> K = k0+0..7,
// h=8..15 -> K = k0+16..23; lanes 16-31 hold the same rows with K phases +8.
// => per lane: two contiguous 16-byte loads.  The B-matrix (KxN) layout is the
// N-mirror of this, so a B-fragment of (KxN) loads with the identical pattern
// from the row-major *transpose* (NxK) storage.
// ---------------------------------------------------------------------------
__device__ __forceinline__ v16h load_frag16(const _Float16* __restrict__ p,
                                            int ld, int row0, int k0) {
  const int lane = threadIdx.x & 31;
  const int r    = row0 + (lane & 15);
  const int kb   = k0 + ((lane >> 4) << 3);       // +0 or +8
  const _Float16* base = p + (size_t)r * ld + kb;
  union { v16h v; f32x4 q[2]; } u;
  u.q[0] = *(const f32x4*)(base);                 // halves 0..7  (K = kb+0..7)
  u.q[1] = *(const f32x4*)(base + 16);            // halves 8..15 (K = kb+16..23)
  return u.v;
}

__device__ __forceinline__ v8f wmma_f16(v16h a, v16h b, v8f c) {
  // D = A(16x32) x B(32x16) + C, f32 accumulate
  return __builtin_amdgcn_wmma_f32_16x16x32_f16(false, a, false, b,
                                                (short)0, c, false, false);
}

// ---------------------------------------------------------------------------
// Kernel 0: transpose weights to NxK row-major and demote to f16.
// ---------------------------------------------------------------------------
__global__ void k_prep_weights(const float* __restrict__ wA,
                               const float* __restrict__ wB,
                               const float* __restrict__ wS,
                               _Float16* __restrict__ wAT,
                               _Float16* __restrict__ wBT,
                               _Float16* __restrict__ wST) {
  const int idx = blockIdx.x * blockDim.x + threadIdx.x;
  if (idx < HH * EE) {
    const int h = idx / EE, e = idx % EE;
    wAT[idx] = (_Float16)wA[e * HH + h];
    wBT[idx] = (_Float16)wB[e * HH + h];
  }
  if (idx < LL * EE) {
    const int l = idx / EE, e = idx % EE;
    wST[idx] = (_Float16)wS[e * LL + l];
  }
}

// ---------------------------------------------------------------------------
// Kernel 1: vemb = char_emb[sent] * gelu_exact(mask_emb[mask]), stored f16.
// ---------------------------------------------------------------------------
__global__ void k_embed(const int* __restrict__ sent, const int* __restrict__ mask,
                        const float* __restrict__ cew, const float* __restrict__ mew,
                        _Float16* __restrict__ vemb) {
  const int idx = blockIdx.x * blockDim.x + threadIdx.x;   // over NTOK*EE = 1M
  if (idx >= NTOK * EE) return;
  const int t = idx >> 8;           // EE = 256
  const int e = idx & (EE - 1);
  const float ce = cew[(size_t)sent[t] * EE + e];
  const float me = mew[(size_t)mask[t] * EE + e];
  const float g  = 0.5f * me * (1.0f + erff(me * 0.70710678118654752f));
  vemb[idx] = (_Float16)(ce * g);
}

// ---------------------------------------------------------------------------
// Kernel 2 (x2): out = act(vemb @ W^T + bias) as f16.
// M=4096, N=512, K=256.  Each wave owns a 32x64 tile: 8 f32 accumulator
// fragments (64 VGPRs) -- fits in registers, no scratch spill.
// ACT: 0 = identity (va), 1 = sigmoid (vb).
// ---------------------------------------------------------------------------
template <int ACT>
__global__ __launch_bounds__(256, 1)
void k_gemm_act(const _Float16* __restrict__ vemb,
                const _Float16* __restrict__ wT,
                const float* __restrict__ bias,
                _Float16* __restrict__ outp) {
  const int wid = (blockIdx.x * blockDim.x + threadIdx.x) >> 5;  // 0..1023
  const int mt = wid >> 3;                                       // 0..127
  const int nt = wid & 7;                                        // 0..7
  const int m0 = mt * 32, n0 = nt * 64;

  v8f c[2][4] = {};

  for (int k = 0; k < EE; k += 32) {
    const v16h a0 = load_frag16(vemb, EE, m0,      k);
    const v16h a1 = load_frag16(vemb, EE, m0 + 16, k);
#pragma unroll
    for (int j = 0; j < 4; ++j) {
      const v16h bf = load_frag16(wT, EE, n0 + 16 * j, k);
      c[0][j] = wmma_f16(a0, bf, c[0][j]);
      c[1][j] = wmma_f16(a1, bf, c[1][j]);
    }
  }

  const int lane = threadIdx.x & 31;
  const int nl   = lane & 15;
  const int mh   = (lane >> 4) * 8;
#pragma unroll
  for (int i = 0; i < 2; ++i) {
#pragma unroll
    for (int j = 0; j < 4; ++j) {
      const int   n = n0 + 16 * j + nl;
      const float b = bias[n];
#pragma unroll
      for (int r = 0; r < 8; ++r) {
        const int m = m0 + 16 * i + mh + r;
        float x = c[i][j][r] + b;
        if (ACT == 1) x = 1.0f / (1.0f + __expf(-x));
        outp[(size_t)m * HH + n] = (_Float16)x;
      }
    }
  }
}

// ---------------------------------------------------------------------------
// Kernel 3: vd = relu(vemb@wS + bS).  M=4096, N=16, K=256.  32x16 per wave.
// ---------------------------------------------------------------------------
__global__ __launch_bounds__(256, 1)
void k_gemm_d(const _Float16* __restrict__ vemb,
              const _Float16* __restrict__ wST,
              const float* __restrict__ bS, float* __restrict__ vd) {
  const int wid = (blockIdx.x * blockDim.x + threadIdx.x) >> 5;  // 0..127
  const int m0 = wid * 32;
  v8f c0 = {}, c1 = {};
  for (int k = 0; k < EE; k += 32) {
    const v16h a0 = load_frag16(vemb, EE, m0,      k);
    const v16h a1 = load_frag16(vemb, EE, m0 + 16, k);
    const v16h bf = load_frag16(wST,  EE, 0,       k);
    c0 = wmma_f16(a0, bf, c0);
    c1 = wmma_f16(a1, bf, c1);
  }
  const int lane = threadIdx.x & 31;
  const int n    = lane & 15;
  const int mh   = (lane >> 4) * 8;
  const float bs = bS[n];
#pragma unroll
  for (int r = 0; r < 8; ++r) {
    const float x0 = c0[r] + bs;
    const float x1 = c1[r] + bs;
    vd[(size_t)(m0 + mh + r)      * LL + n] = x0 > 0.f ? x0 : 0.f;
    vd[(size_t)(m0 + 16 + mh + r) * LL + n] = x1 > 0.f ? x1 : 0.f;
  }
}

// ---------------------------------------------------------------------------
// Kernel 4: vc[b] = va_b @ vb_b^T / H.  Per batch M=N=1024, K=512.
// B = vb^T  =>  B-fragment loads use the A-pattern on row-major vb directly.
// 32x64 tile per wave; 8 v_wmma per K-step, 16 K-steps, accumulators
// stay resident (64 VGPRs).
// ---------------------------------------------------------------------------
__global__ __launch_bounds__(256, 1)
void k_gemm_vc(const _Float16* __restrict__ va,
               const _Float16* __restrict__ vb,
               float* __restrict__ vc) {
  const int wid = (blockIdx.x * blockDim.x + threadIdx.x) >> 5;  // 0..2047
  const int b  = wid >> 9;                                       // /512
  const int t  = wid & 511;
  const int mt = t >> 4;                                         // 0..31
  const int nt = t & 15;                                         // 0..15
  const int i0 = mt * 32, j0 = nt * 64;
  const _Float16* __restrict__ pa = va + (size_t)b * SS * HH;
  const _Float16* __restrict__ pb = vb + (size_t)b * SS * HH;

  v8f c[2][4] = {};

  const int lane = threadIdx.x & 31;
  for (int k = 0; k < HH; k += 32) {
    if (k + 32 < HH) {   // hint next K-slab into cache (global_prefetch_b8)
      __builtin_prefetch(pa + (size_t)(i0 + lane) * HH + k + 32, 0, 3);
      __builtin_prefetch(pb + (size_t)(j0 + lane) * HH + k + 32, 0, 3);
    }
    const v16h a0 = load_frag16(pa, HH, i0,      k);
    const v16h a1 = load_frag16(pa, HH, i0 + 16, k);
#pragma unroll
    for (int j = 0; j < 4; ++j) {
      const v16h bf = load_frag16(pb, HH, j0 + 16 * j, k);
      c[0][j] = wmma_f16(a0, bf, c[0][j]);
      c[1][j] = wmma_f16(a1, bf, c[1][j]);
    }
  }

  const float inv = 1.0f / (float)HH;
  const int nl = lane & 15;
  const int mh = (lane >> 4) * 8;
#pragma unroll
  for (int i = 0; i < 2; ++i) {
#pragma unroll
    for (int j = 0; j < 4; ++j) {
      const int n = j0 + 16 * j + nl;
#pragma unroll
      for (int r = 0; r < 8; ++r) {
        const int m = i0 + 16 * i + mh + r;
        vc[((size_t)b * SS + m) * SS + n] = c[i][j][r] * inv;
      }
    }
  }
}

// ---------------------------------------------------------------------------
// Kernel 5: pred = softmax(vc[...,None] * vd[:,:,None,:], axis=-1).
// Pure streaming: 16 MB read + 256 MB write; one (b,i,j) per thread,
// 64-byte vectorized output.
// ---------------------------------------------------------------------------
__global__ void k_softmax(const float* __restrict__ vc,
                          const float* __restrict__ vd,
                          float* __restrict__ out) {
  const size_t idx = (size_t)blockIdx.x * blockDim.x + threadIdx.x;  // b*S*S
  if (idx >= (size_t)BB * SS * SS) return;
  const float c  = vc[idx];
  const size_t bi = idx >> 10;                  // b*S + i  (S=1024)

  union { f32x4 q[4]; float f[16]; } vdr;
  const f32x4* vp = (const f32x4*)(vd + bi * LL);
  vdr.q[0] = vp[0]; vdr.q[1] = vp[1]; vdr.q[2] = vp[2]; vdr.q[3] = vp[3];

  float x[16];
  float mx = -3.402823466e+38f;
#pragma unroll
  for (int l = 0; l < 16; ++l) { x[l] = c * vdr.f[l]; mx = x[l] > mx ? x[l] : mx; }
  union { f32x4 q[4]; float f[16]; } o;
  float s = 0.f;
#pragma unroll
  for (int l = 0; l < 16; ++l) { const float e = __expf(x[l] - mx); o.f[l] = e; s += e; }
  const float invs = 1.0f / s;
#pragma unroll
  for (int l = 0; l < 16; ++l) o.f[l] *= invs;

  f32x4* op = (f32x4*)(out + idx * LL);
  op[0] = o.q[0]; op[1] = o.q[1]; op[2] = o.q[2]; op[3] = o.q[3];
}

// ---------------------------------------------------------------------------
extern "C" void kernel_launch(void* const* d_in, const int* in_sizes, int n_in,
                              void* d_out, int out_size, void* d_ws, size_t ws_size,
                              hipStream_t stream) {
  (void)in_sizes; (void)n_in; (void)out_size; (void)ws_size;
  const int*   sent = (const int*)d_in[0];
  const int*   mask = (const int*)d_in[1];
  const float* cew  = (const float*)d_in[2];
  const float* mew  = (const float*)d_in[3];
  const float* wA   = (const float*)d_in[4];
  const float* bA   = (const float*)d_in[5];
  const float* wB   = (const float*)d_in[6];
  const float* bB   = (const float*)d_in[7];
  const float* wS   = (const float*)d_in[8];
  const float* bS   = (const float*)d_in[9];
  float* out = (float*)d_out;

  char* ws = (char*)d_ws;
  size_t off = 0;
  auto carve = [&](size_t bytes) -> char* {
    char* p = ws + off;
    off += (bytes + 255) & ~(size_t)255;
    return p;
  };
  _Float16* wAT    = (_Float16*)carve((size_t)HH * EE * sizeof(_Float16));
  _Float16* wBT    = (_Float16*)carve((size_t)HH * EE * sizeof(_Float16));
  _Float16* wST    = (_Float16*)carve((size_t)LL * EE * sizeof(_Float16));
  _Float16* vemb16 = (_Float16*)carve((size_t)NTOK * EE * sizeof(_Float16));
  _Float16* va16   = (_Float16*)carve((size_t)NTOK * HH * sizeof(_Float16));
  _Float16* vb16   = (_Float16*)carve((size_t)NTOK * HH * sizeof(_Float16));
  float*    vd     = (float*)   carve((size_t)NTOK * LL * sizeof(float));
  float*    vc     = (float*)   carve((size_t)BB * SS * SS * sizeof(float));

  k_prep_weights<<<(HH * EE + 255) / 256, 256, 0, stream>>>(wA, wB, wS, wAT, wBT, wST);
  k_embed<<<(NTOK * EE + 255) / 256, 256, 0, stream>>>(sent, mask, cew, mew, vemb16);
  k_gemm_act<0><<<128, 256, 0, stream>>>(vemb16, wAT, bA, va16);
  k_gemm_act<1><<<128, 256, 0, stream>>>(vemb16, wBT, bB, vb16);
  k_gemm_d<<<16, 256, 0, stream>>>(vemb16, wST, bS, vd);
  k_gemm_vc<<<256, 256, 0, stream>>>(va16, vb16, vc);
  k_softmax<<<((size_t)BB * SS * SS + 255) / 256, 256, 0, stream>>>(vc, vd, out);
}
